// EfficientCrossAttention_87582973100415
// MI455X (gfx1250) — matrix operand order repaired and checked by reading, not compiled, not run on hardware
//
#include <hip/hip_runtime.h>
#include <hip/hip_bf16.h>

typedef _Float16 v8h  __attribute__((ext_vector_type(8)));
typedef _Float16 v16h __attribute__((ext_vector_type(16)));
typedef float    v8f  __attribute__((ext_vector_type(8)));
typedef int      v4i  __attribute__((ext_vector_type(4)));

#define EPI_SOFTMAX      0
#define EPI_BIAS_F16     1
#define EPI_F32          2
#define EPI_BIAS_ADD_F32 3

// ---- CDNA5 async global->LDS path (ASYNCcnt-tracked), with sync fallback ----
#if defined(__HIP_DEVICE_COMPILE__) && __has_builtin(__builtin_amdgcn_global_load_async_to_lds_b128)
#define HAVE_ASYNC_LDS 1
#define AS1V4I(p) ((__attribute__((address_space(1))) v4i*)(p))
#define AS3V4I(p) ((__attribute__((address_space(3))) v4i*)(p))
#if __has_builtin(__builtin_amdgcn_s_wait_asynccnt)
#define ASYNC_WAIT() __builtin_amdgcn_s_wait_asynccnt(0)
#else
#define ASYNC_WAIT() asm volatile("s_wait_asynccnt 0x0" ::: "memory")
#endif
#else
#define HAVE_ASYNC_LDS 0
#define ASYNC_WAIT() ((void)0)
#endif

// ---------------------------------------------------------------- utilities
__global__ void cast_f16_kernel(const float* __restrict__ in, _Float16* __restrict__ out, long n) {
    long i = (long)blockIdx.x * blockDim.x + threadIdx.x;
    long stride = (long)gridDim.x * blockDim.x;
    for (; i < n; i += stride) out[i] = (_Float16)in[i];
}

__global__ __launch_bounds__(256) void layernorm_f16_kernel(
    const float* __restrict__ x, const float* __restrict__ g, const float* __restrict__ bta,
    _Float16* __restrict__ out, int C)
{
    __shared__ float s1[256], s2[256];
    const long row = blockIdx.x;
    const float* xr = x + row * C;
    const int tid = threadIdx.x;
    float s = 0.f, ss = 0.f;
    for (int i = tid; i < C; i += 256) { float v = xr[i]; s += v; ss += v * v; }
    s1[tid] = s; s2[tid] = ss; __syncthreads();
    for (int o = 128; o > 0; o >>= 1) {
        if (tid < o) { s1[tid] += s1[tid + o]; s2[tid] += s2[tid + o]; }
        __syncthreads();
    }
    const float mu  = s1[0] / C;
    const float var = s2[0] / C - mu * mu;
    const float rs  = rsqrtf(var + 1e-5f);
    _Float16* orow = out + row * C;
    for (int i = tid; i < C; i += 256) orow[i] = (_Float16)((xr[i] - mu) * rs * g[i] + bta[i]);
}

// ---------------------------------------------------------------- WMMA GEMM
// C[M,N] = A[M,K](f16,row-major,lda) @ B[K,N](f16,row-major,ldb), f32 accum.
// Block tile 128x64x32, 256 threads = 8 waves (4x2 wave grid), 2x2 WMMA tiles/wave.
// Double-buffered LDS; A tile staged via async global->LDS when available.
template <int EPI>
__global__ __launch_bounds__(256) void gemm_wmma_kernel(
    const _Float16* __restrict__ A, int lda, long sAb, long sAh,
    const _Float16* __restrict__ Bw, int ldb, long sBz,
    const float* __restrict__ bias,
    const float* __restrict__ addY, int ldy,
    void* __restrict__ Out, int ldo, long sOb, long sOh,
    int K, int nH)
{
    __shared__ _Float16 As[2][128 * 40];   // pitch 40 halves = 80B (16B multiple)
    __shared__ _Float16 Bs[2][64 * 40];    // stored transposed: [n][k]
    __shared__ float    Cs[(EPI == EPI_SOFTMAX) ? 128 * 65 : 1];

    const int tid  = threadIdx.x;
    const int lane = tid & 31;
    const int w    = tid >> 5;
    const int wm   = w >> 1, wn = w & 1;
    const int hi   = lane >> 4, ln = lane & 15;

    const int z  = blockIdx.z;
    const int bb = z / nH, hh = z - bb * nH;
    const _Float16* Ap = A + (size_t)bb * sAb + (size_t)hh * sAh;
    const _Float16* Bp = Bw + (size_t)z * sBz;
    const long row0 = (long)blockIdx.y * 128;
    const int  col0 = blockIdx.x * 64;

    const int am = tid >> 1;              // 0..127  (A tile row)
    const int ak = (tid & 1) << 4;        // 0 or 16 (A tile k-chunk)
    const int bk = tid >> 3;              // 0..31   (B tile k row)
    const int bn = (tid & 7) << 3;        // 0..56   (B tile n chunk)

    v8f acc[2][2] = {{{}, {}}, {{}, {}}};
    const int nIter = K >> 5;

    // ---------------- prologue: stage tiles for iteration 0
    {
        const _Float16* gA = Ap + (row0 + am) * (long)lda + ak;
#if HAVE_ASYNC_LDS
        _Float16* lA = &As[0][am * 40 + ak];
        __builtin_amdgcn_global_load_async_to_lds_b128(AS1V4I(gA), AS3V4I(lA), 0, 0);
        __builtin_amdgcn_global_load_async_to_lds_b128(AS1V4I(gA), AS3V4I(lA), 16, 0);
#else
        v8h a0 = ((const v8h*)gA)[0];
        v8h a1 = ((const v8h*)gA)[1];
        *(v8h*)&As[0][am * 40 + ak]     = a0;
        *(v8h*)&As[0][am * 40 + ak + 8] = a1;
#endif
        v8h b0 = *(const v8h*)(Bp + (size_t)bk * ldb + (col0 + bn));
#pragma unroll
        for (int j = 0; j < 8; ++j) Bs[0][(bn + j) * 40 + bk] = b0[j];
    }
    ASYNC_WAIT();
    __syncthreads();

    for (int it = 0; it < nIter; ++it) {
        const int cur = it & 1;
        const int nxt = cur ^ 1;
        const bool hasNext = (it + 1) < nIter;
        v8h bstage = {};
#if !HAVE_ASYNC_LDS
        v8h astage0 = {}, astage1 = {};
#endif
        if (hasNext) {
            const int k0n = (it + 1) << 5;
            const _Float16* gA = Ap + (row0 + am) * (long)lda + (k0n + ak);
#if HAVE_ASYNC_LDS
            _Float16* lA = &As[nxt][am * 40 + ak];
            __builtin_amdgcn_global_load_async_to_lds_b128(AS1V4I(gA), AS3V4I(lA), 0, 0);
            __builtin_amdgcn_global_load_async_to_lds_b128(AS1V4I(gA), AS3V4I(lA), 16, 0);
#else
            astage0 = ((const v8h*)gA)[0];
            astage1 = ((const v8h*)gA)[1];
            __builtin_prefetch(gA + 32, 0, 0);
#endif
            bstage = *(const v8h*)(Bp + (size_t)(k0n + bk) * ldb + (col0 + bn));
        }

        // build fragments per CDNA5 16-bit WMMA layouts and multiply
        v16h af[2], bf[2];
#pragma unroll
        for (int i = 0; i < 2; ++i) {
            const int mr = (wm * 32 + i * 16 + ln) * 40;
            v8h lo = *(const v8h*)&As[cur][mr + hi * 8];          // K = hi*8 .. +7
            v8h hv = *(const v8h*)&As[cur][mr + 16 + hi * 8];     // K = 16+hi*8 .. +7
            af[i] = __builtin_shufflevector(lo, hv, 0,1,2,3,4,5,6,7,8,9,10,11,12,13,14,15);
        }
#pragma unroll
        for (int j = 0; j < 2; ++j) {
            const int nr = (wn * 32 + j * 16 + ln) * 40;
            v8h lo = *(const v8h*)&Bs[cur][nr + hi * 8];          // K = hi*8 .. +7
            v8h hv = *(const v8h*)&Bs[cur][nr + 16 + hi * 8];     // K = 16+hi*8 .. +7
            bf[j] = __builtin_shufflevector(lo, hv, 0,1,2,3,4,5,6,7,8,9,10,11,12,13,14,15);
        }
#pragma unroll
        for (int i = 0; i < 2; ++i)
#pragma unroll
            for (int j = 0; j < 2; ++j)
                acc[i][j] = __builtin_amdgcn_wmma_f32_16x16x32_f16(
                    false, af[i], false, bf[j], (short)0, acc[i][j], false, false);

        if (hasNext) {
#if !HAVE_ASYNC_LDS
            *(v8h*)&As[nxt][am * 40 + ak]     = astage0;
            *(v8h*)&As[nxt][am * 40 + ak + 8] = astage1;
#endif
#pragma unroll
            for (int j = 0; j < 8; ++j) Bs[nxt][(bn + j) * 40 + bk] = bstage[j];
        }
        ASYNC_WAIT();
        __syncthreads();
    }

    if constexpr (EPI == EPI_SOFTMAX) {
        // BN==64 aligned blocks => each row's softmax group lives entirely in this block
#pragma unroll
        for (int i = 0; i < 2; ++i)
#pragma unroll
            for (int j = 0; j < 2; ++j) {
                const int cl = wn * 32 + j * 16 + ln;
                const float bval = bias[col0 + cl];
#pragma unroll
                for (int r = 0; r < 8; ++r) {
                    const int rl = wm * 32 + i * 16 + hi * 8 + r;
                    Cs[rl * 65 + cl] = acc[i][j][r] + bval;
                }
            }
        __syncthreads();
        if (tid < 128) {
            float* crow = &Cs[tid * 65];
            float mx = -1e30f;
            for (int j2 = 0; j2 < 64; ++j2) mx = fmaxf(mx, crow[j2]);
            float s = 0.f;
            for (int j2 = 0; j2 < 64; ++j2) { float e = __expf(crow[j2] - mx); crow[j2] = e; s += e; }
            const float inv = 1.0f / s;
            _Float16* op = (_Float16*)Out + (size_t)bb * sOb + (size_t)hh * sOh
                           + (row0 + tid) * (size_t)ldo + col0;
            for (int j2 = 0; j2 < 64; ++j2) op[j2] = (_Float16)(crow[j2] * inv);
        }
    } else {
#pragma unroll
        for (int i = 0; i < 2; ++i)
#pragma unroll
            for (int j = 0; j < 2; ++j) {
                const int cl  = wn * 32 + j * 16 + ln;
                const int col = col0 + cl;
#pragma unroll
                for (int r = 0; r < 8; ++r) {
                    const long rg = row0 + wm * 32 + i * 16 + hi * 8 + r;
                    const float v = acc[i][j][r];
                    if constexpr (EPI == EPI_BIAS_F16) {
                        ((_Float16*)Out)[(size_t)bb * sOb + (size_t)hh * sOh + rg * (size_t)ldo + col]
                            = (_Float16)(v + bias[col]);
                    } else if constexpr (EPI == EPI_F32) {
                        ((float*)Out)[(size_t)bb * sOb + (size_t)hh * sOh + rg * (size_t)ldo + col] = v;
                    } else { // EPI_BIAS_ADD_F32
                        ((float*)Out)[rg * (size_t)ldo + col]
                            = v + bias[col] + addY[rg * (size_t)ldy + col];
                    }
                }
            }
    }
}

// ------------------------------------------------- attn[b,h] = K^T V (64x64, Kdim=2048)
__global__ __launch_bounds__(256) void attn_kernel(
    const _Float16* __restrict__ kq, const _Float16* __restrict__ vq, _Float16* __restrict__ attn)
{
    __shared__ _Float16 ks[32 * 64], vs[32 * 64];
    const int bh = blockIdx.x;
    const int b = bh >> 4, h = bh & 15;
    const _Float16* kp = kq + (size_t)b * 2048 * 1024 + h * 64;
    const _Float16* vp = vq + (size_t)b * 2048 * 1024 + h * 64;
    const int tid = threadIdx.x;
    const int nn = tid >> 3;          // 0..31 tile row to load
    const int cc = (tid & 7) * 8;     // 0..56 column chunk
    const int d  = tid >> 2;          // 0..63 output row
    const int l0 = (tid & 3) * 16;    // output column base
    float acc[16] = {};
    for (int n0 = 0; n0 < 2048; n0 += 32) {
        *(v8h*)&ks[nn * 64 + cc] = *(const v8h*)(kp + (size_t)(n0 + nn) * 1024 + cc);
        *(v8h*)&vs[nn * 64 + cc] = *(const v8h*)(vp + (size_t)(n0 + nn) * 1024 + cc);
        __syncthreads();
#pragma unroll 2
        for (int n = 0; n < 32; ++n) {
            const float kvv = (float)ks[n * 64 + d];
#pragma unroll
            for (int j = 0; j < 16; ++j) acc[j] += kvv * (float)vs[n * 64 + l0 + j];
        }
        __syncthreads();
    }
    _Float16* op = attn + (size_t)bh * 4096 + d * 64 + l0;
#pragma unroll
    for (int j = 0; j < 16; ++j) op[j] = (_Float16)acc[j];
}

// ---------------------------------------------------------------- launcher
extern "C" void kernel_launch(void* const* d_in, const int* in_sizes, int n_in,
                              void* d_out, int out_size, void* d_ws, size_t ws_size,
                              hipStream_t stream) {
    (void)in_sizes; (void)n_in; (void)out_size; (void)ws_size;
    const float* x1 = (const float*)d_in[0];
    const float* x2 = (const float*)d_in[1];
    const float* Wq = (const float*)d_in[2];
    const float* bq = (const float*)d_in[3];
    const float* Wk = (const float*)d_in[4];
    const float* bk = (const float*)d_in[5];
    const float* Wv = (const float*)d_in[6];
    const float* bv = (const float*)d_in[7];
    const float* Wh = (const float*)d_in[8];
    const float* bh = (const float*)d_in[9];
    const float* g1 = (const float*)d_in[10];
    const float* b1 = (const float*)d_in[11];
    const float* g2 = (const float*)d_in[12];
    const float* b2 = (const float*)d_in[13];
    float* out = (float*)d_out;

    constexpr long BT = 4L * 4096;   // 16384 rows of x1
    constexpr long BN = 4L * 2048;   // 8192 rows of x2
    constexpr int  D = 1024, L = 768, H = 16;

    char* base = (char*)d_ws;
    size_t off = 0;
    auto alloc = [&](size_t bytes) -> void* {
        void* p = base + off;
        off = (off + bytes + 255) & ~(size_t)255;
        return p;
    };
    _Float16* nx1 = (_Float16*)alloc(BT * D * 2);
    _Float16* nx2 = (_Float16*)alloc(BN * L * 2);
    _Float16* x1h = (_Float16*)alloc(BT * D * 2);
    _Float16* Wqh = (_Float16*)alloc((long)D * D * 2);
    _Float16* Wkh = (_Float16*)alloc((long)L * D * 2);
    _Float16* Wvh = (_Float16*)alloc((long)L * D * 2);
    _Float16* Whh = (_Float16*)alloc((long)D * D * 2);
    _Float16* q   = (_Float16*)alloc(BT * D * 2);
    _Float16* kk  = (_Float16*)alloc(BN * D * 2);
    _Float16* vv  = (_Float16*)alloc(BN * D * 2);
    _Float16* at  = (_Float16*)alloc(64L * 64 * 64 * 2);
    float*    yws = (float*)alloc(BT * D * 4);

    // f16 conversions
    cast_f16_kernel<<<1024, 256, 0, stream>>>(Wq, Wqh, (long)D * D);
    cast_f16_kernel<<<1024, 256, 0, stream>>>(Wk, Wkh, (long)L * D);
    cast_f16_kernel<<<1024, 256, 0, stream>>>(Wv, Wvh, (long)L * D);
    cast_f16_kernel<<<1024, 256, 0, stream>>>(Wh, Whh, (long)D * D);
    cast_f16_kernel<<<4096, 256, 0, stream>>>(x1, x1h, BT * (long)D);

    // layernorms
    layernorm_f16_kernel<<<(int)BT, 256, 0, stream>>>(x1, g1, b1, nx1, D);
    layernorm_f16_kernel<<<(int)BN, 256, 0, stream>>>(x2, g2, b2, nx2, L);

    // Q = softmax64(nx1 @ Wq + bq)
    gemm_wmma_kernel<EPI_SOFTMAX><<<dim3(D / 64, (int)(BT / 128), 1), 256, 0, stream>>>(
        nx1, D, 0, 0, Wqh, D, 0, bq, nullptr, 0, q, D, 0, 0, D, 1);
    // K = softmax64(nx2 @ Wk + bk)
    gemm_wmma_kernel<EPI_SOFTMAX><<<dim3(D / 64, (int)(BN / 128), 1), 256, 0, stream>>>(
        nx2, L, 0, 0, Wkh, D, 0, bk, nullptr, 0, kk, D, 0, 0, L, 1);
    // V = nx2 @ Wv + bv
    gemm_wmma_kernel<EPI_BIAS_F16><<<dim3(D / 64, (int)(BN / 128), 1), 256, 0, stream>>>(
        nx2, L, 0, 0, Wvh, D, 0, bv, nullptr, 0, vv, D, 0, 0, L, 1);
    // attn[b,h] = K^T V
    attn_kernel<<<64, 256, 0, stream>>>(kk, vv, at);
    // y[b,:,h*64:+64] = q[b,:,h*64:+64] @ attn[b,h]   (batched over 64 heads)
    gemm_wmma_kernel<EPI_F32><<<dim3(1, 4096 / 128, 64), 256, 0, stream>>>(
        q, D, (long)4096 * D, 64, at, 64, 64L * 64, nullptr, nullptr, 0,
        yws, D, (long)4096 * D, 64, 64, H);
    // out = x1 @ Wh + bh + y
    gemm_wmma_kernel<EPI_BIAS_ADD_F32><<<dim3(D / 64, (int)(BT / 128), 1), 256, 0, stream>>>(
        x1h, D, 0, 0, Whh, D, 0, bh, yws, D, out, D, 0, 0, D, 1);
}
